// TennisMatchPredictor_87101936763491
// MI455X (gfx1250) — compile-verified, matrix operand order checked by reading, not run on hardware
//
#include <hip/hip_runtime.h>

typedef __attribute__((ext_vector_type(16))) _Float16 v16h;
typedef __attribute__((ext_vector_type(8)))  float    v8f;

#define BATCH   16384
#define TLEN    128
#define HGRU    8
#define NGRU    4
#define GROUP   16      // batch rows handled per wave
#define LPAD    132     // LDS row stride in floats (16B aligned, bank-spread)

struct GruArgs {
  const float* x[NGRU];
  const float* wih[NGRU];
  const float* whh[NGRU];
  const float* bih[NGRU];
  const float* bhh[NGRU];
  float* hout;          // [NGRU][BATCH][HGRU], tanh already applied
};

// Branch-free fast activations on v_exp_f32 / v_rcp_f32 (no IEEE divide, no
// EXEC-divergent OCML tanh on the recurrence critical path).
__device__ __forceinline__ float sigf(float x) {
  return __builtin_amdgcn_rcpf(1.0f + __expf(-x));
}
__device__ __forceinline__ float tanh_fast(float x) {
  // tanh(x) = 2*sigmoid(2x) - 1; saturates cleanly to +/-1 in f32.
  return fmaf(2.0f, __builtin_amdgcn_rcpf(1.0f + __expf(-2.0f * x)), -1.0f);
}

// One wave = one GRU over 16 batch rows. Recurrence as three independent
// D[8x16] = W_gate[8x8] x h^T tiles per step via v_wmma_f32_16x16x32_f16
// (K padded 8->32, M padded 8->16 with zeros). Putting r/z/n gate rows at
// M=0..7 of separate tiles means every lane l<16 receives ALL gate
// pre-activations for batch l in its own accumulator VGPRs -> no cross-lane
// shuffle (no ds_bpermute / dscnt wait) on the sequential critical path.
__launch_bounds__(128, 1)
__global__ void gru_wmma_kernel(GruArgs a) {
  __shared__ __align__(16) float xs[NGRU][GROUP][LPAD];
  const int lane = threadIdx.x & 31;
  const int w    = threadIdx.x >> 5;       // which GRU (0..3)
  const int b0   = blockIdx.x * GROUP;

  // Stage this wave's 16x128 sequence block into LDS (coalesced b128 loads).
  const float* xsrc = a.x[w];
  for (int r = 0; r < GROUP; ++r) {
    const float4 v = reinterpret_cast<const float4*>(xsrc + (size_t)(b0 + r) * TLEN)[lane];
    *reinterpret_cast<float4*>(&xs[w][r][lane * 4]) = v;
  }
  __syncthreads();

  const float* Wih = a.wih[w];   // [24,1]
  const float* Whh = a.whh[w];   // [24,8] row-major
  const float* bih = a.bih[w];   // [24]
  const float* bhh = a.bhh[w];   // [24]

  // Loop-invariant A fragments (16-bit A 16x32 layout: lanes 0-15 hold K=0..7
  // in elements 0..7). Gate rows live at M=0..7 only; everything else zero.
  v16h A1 = {};  // Whh rows  0..7  (r gates)
  v16h A3 = {};  // Whh rows  8..15 (z gates)
  v16h A2 = {};  // Whh rows 16..23 (n gates)
  if (lane < 8) {
#pragma unroll
    for (int e = 0; e < 8; ++e) {
      A1[e] = (_Float16)Whh[lane * 8 + e];
      A3[e] = (_Float16)Whh[(8 + lane) * 8 + e];
      A2[e] = (_Float16)Whh[(16 + lane) * 8 + e];
    }
  }

  // Wave-uniform C-tile coefficients (only lanes 0-15 of D matter now).
  // C(m=v, n=batch): r/z tiles get x_t*Wih + bih + bhh; n tile gets bhh_n.
  float swr[8], sbr[8], swz[8], sbz[8], sbhn[8], swn[8], sbn[8];
#pragma unroll
  for (int v = 0; v < 8; ++v) {
    swr[v]  = Wih[v];       sbr[v] = bih[v] + bhh[v];
    swz[v]  = Wih[8 + v];   sbz[v] = bih[8 + v] + bhh[8 + v];
    sbhn[v] = bhh[16 + v];
    swn[v]  = Wih[16 + v];  sbn[v] = bih[16 + v];
  }

  float h[8];
#pragma unroll
  for (int v = 0; v < 8; ++v) h[v] = 0.0f;
  v16h Bf = {};                       // h^T fragment (f16), zero-padded K>=8

  const float* xrow = xs[w][lane & 15];

  // 4 time steps per ds_load_b128, software-pipelined one chunk ahead
  // (consumed in reverse = jnp.flip order).
  float4 xq = *reinterpret_cast<const float4*>(&xrow[TLEN - 4]);
  for (int sg = 0; sg < TLEN / 4; ++sg) {
    const float4 xq_cur = xq;
    const int nxt = (sg + 1) & (TLEN / 4 - 1);          // wraps harmlessly
    xq = *reinterpret_cast<const float4*>(&xrow[TLEN - 4 - nxt * 4]);

    const float xts[4] = { xq_cur.w, xq_cur.z, xq_cur.y, xq_cur.x };
#pragma unroll
    for (int u = 0; u < 4; ++u) {
      const float xt = xts[u];

      v8f c1, c2, c3;
#pragma unroll
      for (int v = 0; v < 8; ++v) {
        c1[v] = fmaf(xt, swr[v], sbr[v]);
        c3[v] = fmaf(xt, swz[v], sbz[v]);
        c2[v] = sbhn[v];
      }

      // Three independent tiles, all consuming the same B fragment:
      // d1 = r pre-acts, d3 = z pre-acts, d2 = gh_n + bhh_n  (lanes 0-15).
      v8f d1 = __builtin_amdgcn_wmma_f32_16x16x32_f16(false, A1, false, Bf,
                                                      (short)0, c1, false, false);
      v8f d3 = __builtin_amdgcn_wmma_f32_16x16x32_f16(false, A3, false, Bf,
                                                      (short)0, c3, false, false);
      v8f d2 = __builtin_amdgcn_wmma_f32_16x16x32_f16(false, A2, false, Bf,
                                                      (short)0, c2, false, false);

#pragma unroll
      for (int v = 0; v < 8; ++v) {
        const float r  = sigf(d1[v]);
        const float z  = sigf(d3[v]);
        const float gn = fmaf(xt, swn[v], sbn[v]);
        const float nn = tanh_fast(fmaf(r, d2[v], gn));
        h[v] = fmaf(z, h[v] - nn, nn);           // (1-z)*n + z*h
      }
      // Rebuild B fragment: lane l<16 holds h[l][0..7] packed f16; rest zero.
#pragma unroll
      for (int e = 0; e < 8; ++e)
        Bf[e] = (lane < 16) ? (_Float16)h[e] : (_Float16)0.0f;
    }
  }

  if (lane < 16) {
    float* o = a.hout + ((size_t)w * BATCH + b0 + lane) * HGRU;
#pragma unroll
    for (int v = 0; v < 8; ++v) o[v] = tanh_fast(h[v]);
  }
}

// Player towers + concat + final MLP; one thread per batch row (weights uniform).
__launch_bounds__(256, 1)
__global__ void head_kernel(const float* __restrict__ tf,
                            const float* __restrict__ p1, const float* __restrict__ p2,
                            const float* __restrict__ Wp1, const float* __restrict__ bp1,
                            const float* __restrict__ Wp1b, const float* __restrict__ bp1b,
                            const float* __restrict__ Wp2, const float* __restrict__ bp2,
                            const float* __restrict__ Wp2b, const float* __restrict__ bp2b,
                            const float* __restrict__ W1, const float* __restrict__ b1,
                            const float* __restrict__ W2, const float* __restrict__ b2,
                            const float* __restrict__ hout, float* __restrict__ out) {
  const int b = blockIdx.x * blockDim.x + threadIdx.x;
  if (b >= BATCH) return;

  float feat[66];
  feat[0] = tf[b * 2 + 0];
  feat[1] = tf[b * 2 + 1];

  const float* pf[2] = { p1 + (size_t)b * 64, p2 + (size_t)b * 64 };
  const float* WA[2] = { Wp1, Wp2 };   const float* bA[2] = { bp1, bp2 };
  const float* WB[2] = { Wp1b, Wp2b }; const float* bB[2] = { bp1b, bp2b };
  for (int pl = 0; pl < 2; ++pl) {
    float hp[16];
    for (int i = 0; i < 16; ++i) {
      float acc = bA[pl][i];
      const float* wr = WA[pl] + i * 64;
      for (int k = 0; k < 64; ++k) acc = fmaf(pf[pl][k], wr[k], acc);
      hp[i] = sigf(acc);
    }
    for (int i = 0; i < 16; ++i) {
      float acc = bB[pl][i];
      const float* wr = WB[pl] + i * 16;
#pragma unroll
      for (int k = 0; k < 16; ++k) acc = fmaf(hp[k], wr[k], acc);
      feat[2 + pl * 16 + i] = tanh_fast(acc);
    }
  }
  for (int g = 0; g < 4; ++g) {
    const float* hs = hout + ((size_t)g * BATCH + b) * HGRU;
#pragma unroll
    for (int j = 0; j < 8; ++j) feat[34 + g * 8 + j] = hs[j];
  }

  float hh[16];
  for (int i = 0; i < 16; ++i) {
    float acc = b1[i];
    const float* wr = W1 + i * 66;
    for (int k = 0; k < 66; ++k) acc = fmaf(feat[k], wr[k], acc);
    hh[i] = tanh_fast(acc);
  }
  float acc = b2[0];
#pragma unroll
  for (int i = 0; i < 16; ++i) acc = fmaf(hh[i], W2[i], acc);
  out[b] = tanh_fast(acc);
}

extern "C" void kernel_launch(void* const* d_in, const int* in_sizes, int n_in,
                              void* d_out, int out_size, void* d_ws, size_t ws_size,
                              hipStream_t stream) {
  (void)in_sizes; (void)n_in; (void)out_size; (void)ws_size;

  // setup_inputs() order:
  // 0 tf, 1 p1, 2 p2, 3 h2h_overall, 4 h2h_surface, 5 shape_p1, 6 shape_p2,
  // 7-10 masks (identity in reference -> skipped),
  // 11 Wp1, 12 bp1, 13 Wp1b, 14 bp1b, 15 Wp2, 16 bp2, 17 Wp2b, 18 bp2b,
  // 19+4g: Wih, 20+4g: Whh, 21+4g: bih, 22+4g: bhh  (g = ho,hs,s1,s2),
  // 35 W1, 36 b1, 37 W2, 38 b2
  GruArgs ga;
  for (int g = 0; g < NGRU; ++g) {
    ga.x[g]   = (const float*)d_in[3 + g];
    ga.wih[g] = (const float*)d_in[19 + 4 * g];
    ga.whh[g] = (const float*)d_in[20 + 4 * g];
    ga.bih[g] = (const float*)d_in[21 + 4 * g];
    ga.bhh[g] = (const float*)d_in[22 + 4 * g];
  }
  float* hout = (float*)d_ws;        // NGRU*BATCH*HGRU floats = 2 MB
  ga.hout = hout;

  gru_wmma_kernel<<<BATCH / GROUP, 128, 0, stream>>>(ga);

  head_kernel<<<BATCH / 256, 256, 0, stream>>>(
      (const float*)d_in[0], (const float*)d_in[1], (const float*)d_in[2],
      (const float*)d_in[11], (const float*)d_in[12], (const float*)d_in[13], (const float*)d_in[14],
      (const float*)d_in[15], (const float*)d_in[16], (const float*)d_in[17], (const float*)d_in[18],
      (const float*)d_in[35], (const float*)d_in[36], (const float*)d_in[37], (const float*)d_in[38],
      hout, (float*)d_out);
}
